// DimeModule_31791347925876
// MI455X (gfx1250) — matrix operand order repaired and checked by reading, not compiled, not run on hardware
//
#include <hip/hip_runtime.h>
#include <stdint.h>

#define DIMD  128
#define NRBF  16
#define NSBF  64
#define NBIL  8
#define NEDGE 100000
#define NTRIP 500000
#define NATOM 10000

typedef __attribute__((ext_vector_type(16))) __bf16 bf16x16;
typedef __attribute__((ext_vector_type(8)))  float  f32x8;

union BFrag { bf16x16 v; uint32_t u[8]; uint4 q[2]; };
union CFrag { f32x8 v; float f[8]; };

// Hardware bf16 conversion: let clang pick v_cvt_pk_bf16_f32 / native cvt.
__device__ __forceinline__ uint32_t f2bf2(float lo, float hi){
  union { __bf16 h[2]; uint32_t u; } p;
  p.h[0] = (__bf16)lo;
  p.h[1] = (__bf16)hi;
  return p.u;
}
__device__ __forceinline__ float bf2f(uint16_t b){
  return __uint_as_float(((uint32_t)b) << 16);
}
__device__ __forceinline__ float silu_f(float x){
  return x / (1.0f + __expf(-x));
}

// ---------------------------------------------------------------------------
// Weight packing: f32 [K, 128] row-major -> bf16 WMMA B-fragment order.
// B (32x16 bf16): lane 0-15 -> N=lane, K={0..7,16..23}; lane 16-31 -> N=lane-16, K={8..15,24..31}.
// Storage: pk[((kt*8 + nt)*32 + lane)*16 + j]
// ---------------------------------------------------------------------------
__global__ void pack_dense_kernel(const float* __restrict__ W, uint16_t* __restrict__ out){
  int idx = blockIdx.x * 256 + threadIdx.x;          // 4*8*32*16 = 16384
  if (idx >= 4*8*32*16) return;
  int j    = idx & 15;
  int lane = (idx >> 4) & 31;
  int nt   = (idx >> 9) & 7;
  int kt   = idx >> 12;
  int k = kt*32 + ((lane < 16) ? 0 : 8) + (j & 7) + ((j & 8) ? 16 : 0);
  int n = nt*16 + (lane & 15);
  ((__bf16*)out)[idx] = (__bf16)W[k*DIMD + n];
}

// W_bi [i, j, l] f32 -> B[k = i*8 + l, n = j], packed as above (kt = 0..31).
__global__ void pack_wbi_kernel(const float* __restrict__ Wbi, uint16_t* __restrict__ out){
  int idx = blockIdx.x * 256 + threadIdx.x;          // 32*8*32*16 = 131072
  if (idx >= 32*8*32*16) return;
  int j    = idx & 15;
  int lane = (idx >> 4) & 31;
  int nt   = (idx >> 9) & 7;
  int kt   = idx >> 12;
  int k = kt*32 + ((lane < 16) ? 0 : 8) + (j & 7) + ((j & 8) ? 16 : 0);        // 0..1023
  int i = k >> 3, l = k & 7;
  int n = nt*16 + (lane & 15);
  ((__bf16*)out)[idx] = (__bf16)Wbi[((size_t)i*DIMD + n)*NBIL + l];
}

// ---------------------------------------------------------------------------
// A fragment (16x32 bf16) for 16 consecutive f32 rows of X[*,128].
// ISA layout: lane 0-15 (M=lane): K = {kk..kk+7, kk+16..kk+23};
//             lane 16-31 (M=lane-16): K = {kk+8..kk+15, kk+24..kk+31}.
// ---------------------------------------------------------------------------
__device__ __forceinline__ bf16x16 build_afrag(const float* __restrict__ Xrow, int kk, int lane){
  int kb = kk + ((lane & 16) ? 8 : 0);
  const float4* p0 = (const float4*)(Xrow + kb);
  const float4* p1 = (const float4*)(Xrow + kb + 16);
  float4 a0 = p0[0], a1 = p0[1];
  float4 b0 = p1[0], b1 = p1[1];
  BFrag r;
  r.u[0] = f2bf2(a0.x, a0.y);  r.u[1] = f2bf2(a0.z, a0.w);
  r.u[2] = f2bf2(a1.x, a1.y);  r.u[3] = f2bf2(a1.z, a1.w);
  r.u[4] = f2bf2(b0.x, b0.y);  r.u[5] = f2bf2(b0.z, b0.w);
  r.u[6] = f2bf2(b1.x, b1.y);  r.u[7] = f2bf2(b1.z, b1.w);
  return r.v;
}

__device__ __forceinline__ bf16x16 load_bfrag(const uint16_t* __restrict__ Wpk, int kt, int nt, int lane){
  const uint4* bp = ((const uint4*)Wpk) + (((size_t)(kt*8 + nt)*32 + lane) << 1);
  BFrag b; b.q[0] = bp[0]; b.q[1] = bp[1];
  return b.v;
}

// ---------------------------------------------------------------------------
// Generic 128-wide linear: Y = (ACT? silu : id)(X @ W [+ bias]) [+ gate*add]
// Compile-time specialization removes all epilogue branches.
// One wave per 16-row M-tile, A-fragments cached in registers across n-tiles.
// ---------------------------------------------------------------------------
template<int ACT, int HAS_BIAS, int HAS_ADD, int HAS_GATE>
__global__ void __launch_bounds__(256) gemm128_t(
    const float* __restrict__ X, const uint16_t* __restrict__ Wpk,
    const float* __restrict__ bias, const float* __restrict__ add,
    const float* __restrict__ gate, float* __restrict__ Y, int M){
  int lane = threadIdx.x & 31;
  int wave = threadIdx.x >> 5;
  int tile = blockIdx.x * 8 + wave;
  if (tile * 16 >= M) return;
  int m0 = tile * 16;
  const float* Xrow = X + (size_t)(m0 + (lane & 15)) * DIMD;
  int mb = m0 + ((lane & 16) ? 8 : 0);
  int nl = lane & 15;

  bf16x16 af[4];
  #pragma unroll
  for (int kt = 0; kt < 4; ++kt) af[kt] = build_afrag(Xrow, kt*32, lane);

  #pragma unroll
  for (int nt = 0; nt < 8; ++nt){
    CFrag c;
    #pragma unroll
    for (int p = 0; p < 8; ++p) c.f[p] = 0.0f;
    #pragma unroll
    for (int kt = 0; kt < 4; ++kt){
      bf16x16 b = load_bfrag(Wpk, kt, nt, lane);
      c.v = __builtin_amdgcn_wmma_f32_16x16x32_bf16(false, af[kt], false, b, (short)0, c.v, false, false);
    }
    int n = nt*16 + nl;
    float bv = HAS_BIAS ? bias[n] : 0.0f;
    float gv = HAS_GATE ? gate[n] : 1.0f;
    #pragma unroll
    for (int p = 0; p < 8; ++p){
      size_t off = (size_t)(mb + p) * DIMD + n;
      float v = c.f[p] + bv;
      if (ACT) v = silu_f(v);
      if (HAS_ADD) v += gv * add[off];
      Y[off] = v;
    }
  }
}

// ---------------------------------------------------------------------------
// rh = (rbf @ W_rbf) * silu(msg @ W_s + b_s), stored as bf16 [E,128]
// ---------------------------------------------------------------------------
__global__ void __launch_bounds__(256) rh_kernel(
    const float* __restrict__ msg, const uint16_t* __restrict__ WsPk,
    const float* __restrict__ bs, const float* __restrict__ rbf,
    const float* __restrict__ Wrbf, uint16_t* __restrict__ rh_bf, int M){
  int lane = threadIdx.x & 31;
  int wave = threadIdx.x >> 5;
  int tile = blockIdx.x * 8 + wave;
  if (tile * 16 >= M) return;
  int m0 = tile * 16;
  const float* Xrow = msg + (size_t)(m0 + (lane & 15)) * DIMD;
  int mb = m0 + ((lane & 16) ? 8 : 0);
  int nl = lane & 15;

  bf16x16 af[4];
  #pragma unroll
  for (int kt = 0; kt < 4; ++kt) af[kt] = build_afrag(Xrow, kt*32, lane);

  #pragma unroll
  for (int nt = 0; nt < 8; ++nt){
    CFrag c;
    #pragma unroll
    for (int p = 0; p < 8; ++p) c.f[p] = 0.0f;
    #pragma unroll
    for (int kt = 0; kt < 4; ++kt){
      bf16x16 b = load_bfrag(WsPk, kt, nt, lane);
      c.v = __builtin_amdgcn_wmma_f32_16x16x32_bf16(false, af[kt], false, b, (short)0, c.v, false, false);
    }
    int n = nt*16 + nl;
    float wcol[NRBF];
    #pragma unroll
    for (int s = 0; s < NRBF; ++s) wcol[s] = Wrbf[s*DIMD + n];
    float bv = bs[n];
    #pragma unroll
    for (int p = 0; p < 8; ++p){
      int m = mb + p;
      float h = silu_f(c.f[p] + bv);
      const float* rrow = rbf + (size_t)m * NRBF;
      float r = 0.0f;
      #pragma unroll
      for (int s = 0; s < NRBF; ++s) r += rrow[s] * wcol[s];
      ((__bf16*)rh_bf)[(size_t)m*DIMD + n] = (__bf16)(r * h);
    }
  }
}

// ---------------------------------------------------------------------------
// Triplet bilinear einsum + scatter-add (persistent blocks).
// msg[w,j] = sum_{k=(i,l)} (rh[src[w],i] * sbfp[w,l]) * Wbi_pk[k,j]
// Each block caches the full packed W_bi (256 KB) in LDS once, then loops over
// 16-triplet tiles: A-tile (16x1024 bf16, 32 KB) staged in LDS in fragment
// order, 8 waves x 32 WMMAs per tile, f32 atomics into aggr[dst].
// Dynamic LDS: 256KB (B) + 32KB (A) + 1KB (sbfp/idx) < 320KB per WGP.
// ---------------------------------------------------------------------------
#define EIN_SMEM (262144 + 32768 + 1024)

__global__ void __launch_bounds__(256) einsum_scatter_kernel(
    const uint16_t* __restrict__ rh_bf, const float* __restrict__ sbf,
    const float* __restrict__ Wsbf, const uint16_t* __restrict__ WbiPk,
    const int* __restrict__ eidx, float* __restrict__ aggr, int numTiles){
  extern __shared__ uint8_t smem[];
  uint16_t* Bs = (uint16_t*)smem;                                   // 256 KB
  uint16_t (*Apk)[32][16] = (uint16_t (*)[32][16])(smem + 262144);  // 32 KB
  float* sbfp = (float*)(smem + 262144 + 32768);                    // 16*8 f32
  int*   srcI = (int*)(sbfp + 16*NBIL);
  int*   dstI = srcI + 16;

  int tid = threadIdx.x;

  // Cache packed W_bi in LDS (16384 uint4 / 256 threads = 64 each)
  for (int it = tid; it < 16384; it += 256)
    ((uint4*)Bs)[it] = ((const uint4*)WbiPk)[it];

  int lane = tid & 31, nt = tid >> 5;
  int n  = nt*16 + (lane & 15);
  int mbh = (lane & 16) ? 8 : 0;

  for (int tile = blockIdx.x; tile < numTiles; tile += gridDim.x){
    __syncthreads();   // previous tile fully consumed (and Bs ready on iter 0)
    int w0 = tile * 16;
    if (tid < 128){                       // sbf projection 64 -> 8
      int r = tid >> 3, l = tid & 7;
      const float* srow = sbf + (size_t)(w0 + r) * NSBF;
      float acc = 0.0f;
      #pragma unroll
      for (int s = 0; s < NSBF; ++s) acc += srow[s] * Wsbf[s*NBIL + l];
      sbfp[r*NBIL + l] = acc;
    } else if (tid < 144){
      int r = tid - 128;
      srcI[r] = eidx[w0 + r];             // row 0: src
      dstI[r] = eidx[NTRIP + w0 + r];     // row 1: dst
    }
    __syncthreads();

    // Build A tile: thread -> (m = tid>>4, i = (tid&15)*8 .. +7)
    {
      int m  = tid >> 4;
      int ib = (tid & 15) * 8;
      const uint16_t* rrow = rh_bf + (size_t)srcI[m] * DIMD;
      uint4 rv = *(const uint4*)(rrow + ib);           // 8 gathered bf16
      const uint16_t* rv16 = (const uint16_t*)&rv;
      float s0 = sbfp[m*NBIL+0], s1 = sbfp[m*NBIL+1], s2 = sbfp[m*NBIL+2], s3 = sbfp[m*NBIL+3];
      float s4 = sbfp[m*NBIL+4], s5 = sbfp[m*NBIL+5], s6 = sbfp[m*NBIL+6], s7 = sbfp[m*NBIL+7];
      #pragma unroll
      for (int t = 0; t < 8; ++t){
        int i = ib + t;
        float rvf = bf2f(rv16[t]);
        int alane = (tid >> 4) + 16 * (i & 1);
        int j0    = ((i >> 1) & 1) * 8;
        int kt    = i >> 2;
        uint4 w4;
        w4.x = f2bf2(rvf * s0, rvf * s1);
        w4.y = f2bf2(rvf * s2, rvf * s3);
        w4.z = f2bf2(rvf * s4, rvf * s5);
        w4.w = f2bf2(rvf * s6, rvf * s7);
        *(uint4*)&Apk[kt][alane][j0] = w4;
      }
    }
    __syncthreads();

    CFrag c;
    #pragma unroll
    for (int p = 0; p < 8; ++p) c.f[p] = 0.0f;
    #pragma unroll 8
    for (int kt = 0; kt < 32; ++kt){
      BFrag a;
      const uint4* ap = (const uint4*)&Apk[kt][lane][0];
      a.q[0] = ap[0]; a.q[1] = ap[1];
      BFrag b;
      const uint4* bp = ((const uint4*)Bs) + (((kt*8 + nt)*32 + lane) << 1);
      b.q[0] = bp[0]; b.q[1] = bp[1];
      c.v = __builtin_amdgcn_wmma_f32_16x16x32_bf16(false, a.v, false, b.v, (short)0, c.v, false, false);
    }
    #pragma unroll
    for (int p = 0; p < 8; ++p){
      int d = dstI[mbh + p];
      unsafeAtomicAdd(&aggr[(size_t)d*DIMD + n], c.f[p]);
    }
  }
}

// atom[aidx[e], n] += (rbf[e] @ Worbf)[n] * mji[e, n]
__global__ void out_scatter_kernel(
    const float* __restrict__ mji, const float* __restrict__ rbf,
    const float* __restrict__ Worbf, const int* __restrict__ aidx,
    float* __restrict__ atomv){
  int gid = blockIdx.x * 256 + threadIdx.x;
  int e = gid >> 7, n = gid & 127;
  if (e >= NEDGE) return;
  const float* rrow = rbf + (size_t)e * NRBF;
  float g = 0.0f;
  #pragma unroll
  for (int s = 0; s < NRBF; ++s) g += rrow[s] * Worbf[s*DIMD + n];
  unsafeAtomicAdd(&atomv[(size_t)aidx[e]*DIMD + n], g * mji[(size_t)e*DIMD + n]);
}

__global__ void final_out_kernel(const float* __restrict__ a, const float* __restrict__ Wf,
                                 float* __restrict__ out){
  int i = blockIdx.x * 256 + threadIdx.x;
  if (i >= NATOM) return;
  const float* row = a + (size_t)i * DIMD;
  float acc = 0.0f;
  #pragma unroll 4
  for (int n = 0; n < DIMD; ++n) acc += row[n] * Wf[n];
  out[i] = acc;
}

// ---------------------------------------------------------------------------
extern "C" void kernel_launch(void* const* d_in, const int* in_sizes, int n_in,
                              void* d_out, int out_size, void* d_ws, size_t ws_size,
                              hipStream_t stream){
  (void)in_sizes; (void)n_in; (void)out_size; (void)ws_size;
  // setup_inputs order, params flattened as JAX pytree (dict keys sorted)
  const float* msg   = (const float*)d_in[0];
  const float* rbf   = (const float*)d_in[1];
  const float* sbf   = (const float*)d_in[2];
  const int*   eidx  = (const int*)d_in[3];
  const int*   aidx  = (const int*)d_in[4];
  const float* Wbi   = (const float*)d_in[5];   // W_bi [128,128,8]
  const float* gate  = (const float*)d_in[6];   // gate [1,128]
  const float* WintW = (const float*)d_in[7];   // lin_interact.W
  const float* Wintb = (const float*)d_in[8];
  const float* Wrbf  = (const float*)d_in[9];   // lin_rbf.W [16,128]
  const float* WsW   = (const float*)d_in[10];  // lin_s.W
  const float* Wsb   = (const float*)d_in[11];
  const float* Wsbf  = (const float*)d_in[12];  // lin_sbf.W [64,8]
  const float* WtW   = (const float*)d_in[13];  // lin_t.W
  const float* Wtb   = (const float*)d_in[14];
  const float* Wfin  = (const float*)d_in[15];  // out_final.W [128,1]
  const float* Worbf = (const float*)d_in[16];  // out_lin_rbf.W [16,128]
  const float* Wo0   = (const float*)d_in[17];
  const float* Wo0b  = (const float*)d_in[18];
  const float* Wo1   = (const float*)d_in[19];
  const float* Wo1b  = (const float*)d_in[20];
  const float* Wo2   = (const float*)d_in[21];
  const float* Wo2b  = (const float*)d_in[22];
  const float* Ri1W  = (const float*)d_in[23];
  const float* Ri1b  = (const float*)d_in[24];
  const float* Ri2W  = (const float*)d_in[25];
  const float* Ri2b  = (const float*)d_in[26];
  const float* Rm01W = (const float*)d_in[27];
  const float* Rm01b = (const float*)d_in[28];
  const float* Rm02W = (const float*)d_in[29];
  const float* Rm02b = (const float*)d_in[30];
  const float* Rm11W = (const float*)d_in[31];
  const float* Rm11b = (const float*)d_in[32];
  const float* Rm12W = (const float*)d_in[33];
  const float* Rm12b = (const float*)d_in[34];

  // workspace layout
  uint8_t* ws = (uint8_t*)d_ws;
  size_t off = 0;
  auto carve = [&](size_t bytes) -> uint8_t* {
    uint8_t* p = ws + off;
    off = (off + bytes + 255) & ~(size_t)255;
    return p;
  };
  uint16_t* rh_bf = (uint16_t*)carve((size_t)NEDGE * DIMD * 2);
  float*    bufA  = (float*)   carve((size_t)NEDGE * DIMD * 4);  // silu(lin_t)+aggr, res chain
  float*    bufZ  = (float*)   carve((size_t)NEDGE * DIMD * 4);
  float*    atomA = (float*)   carve((size_t)NATOM * DIMD * 4);
  float*    atomB = (float*)   carve((size_t)NATOM * DIMD * 4);
  uint16_t* wbiPk = (uint16_t*)carve(32*8*32*16 * 2);
  uint16_t* dpk[12];
  for (int i = 0; i < 12; ++i) dpk[i] = (uint16_t*)carve(4*8*32*16 * 2);

  float* mji_out = (float*)d_out;
  float* out_v   = mji_out + (size_t)NEDGE * DIMD;

  hipMemsetAsync(atomA, 0, (size_t)NATOM * DIMD * 4, stream);

  // pack weights to bf16 fragment layout
  pack_wbi_kernel<<<512, 256, 0, stream>>>(Wbi, wbiPk);
  const float* dsrc[12] = {WsW, WtW, WintW, Ri1W, Ri2W, Rm01W, Rm02W, Rm11W, Rm12W, Wo0, Wo1, Wo2};
  for (int i = 0; i < 12; ++i)
    pack_dense_kernel<<<64, 256, 0, stream>>>(dsrc[i], dpk[i]);

  auto blocks_of = [](int M){ return (M/16 + 7) / 8; };
  const int EB = blocks_of(NEDGE);   // 782

  // rh = (rbf @ Wrbf) * silu(msg @ Ws + bs)  -> bf16
  rh_kernel<<<EB, 256, 0, stream>>>(msg, dpk[0], Wsb, rbf, Wrbf, rh_bf, NEDGE);
  // bufA = silu(msg @ Wt + bt)   (einsum atomics accumulate directly on top)
  gemm128_t<1,1,0,0><<<EB, 256, 0, stream>>>(msg, dpk[1], Wtb, nullptr, nullptr, bufA, NEDGE);
  // bufA += einsum scatter (persistent blocks, W_bi cached in LDS)
  einsum_scatter_kernel<<<2048, 256, EIN_SMEM, stream>>>(rh_bf, sbf, Wsbf, wbiPk, eidx, bufA, NTRIP/16);
  // res_inter: bufA = bufA + silu(l2(silu(l1(bufA))))
  gemm128_t<1,1,0,0><<<EB, 256, 0, stream>>>(bufA, dpk[3], Ri1b, nullptr, nullptr, bufZ, NEDGE);
  gemm128_t<1,1,1,0><<<EB, 256, 0, stream>>>(bufZ, dpk[4], Ri2b, bufA, nullptr, bufA, NEDGE);
  // interact: bufZ = silu(lin_int(bufA)) + gate * msg
  gemm128_t<1,1,1,1><<<EB, 256, 0, stream>>>(bufA, dpk[2], Wintb, msg, gate, bufZ, NEDGE);
  // res_msg 0: bufZ = bufZ + silu(l2(silu(l1(bufZ))))
  gemm128_t<1,1,0,0><<<EB, 256, 0, stream>>>(bufZ, dpk[5], Rm01b, nullptr, nullptr, bufA, NEDGE);
  gemm128_t<1,1,1,0><<<EB, 256, 0, stream>>>(bufA, dpk[6], Rm02b, bufZ, nullptr, bufZ, NEDGE);
  // res_msg 1 -> mji output
  gemm128_t<1,1,0,0><<<EB, 256, 0, stream>>>(bufZ, dpk[7], Rm11b, nullptr, nullptr, bufA, NEDGE);
  gemm128_t<1,1,1,0><<<EB, 256, 0, stream>>>(bufA, dpk[8], Rm12b, bufZ, nullptr, mji_out, NEDGE);
  // output block
  out_scatter_kernel<<<(NEDGE*DIMD)/256, 256, 0, stream>>>(mji_out, rbf, Worbf, aidx, atomA);
  const int AB = blocks_of(NATOM);   // 79
  gemm128_t<1,1,0,0><<<AB, 256, 0, stream>>>(atomA, dpk[9],  Wo0b, nullptr, nullptr, atomB, NATOM);
  gemm128_t<1,1,0,0><<<AB, 256, 0, stream>>>(atomB, dpk[10], Wo1b, nullptr, nullptr, atomA, NATOM);
  gemm128_t<1,1,0,0><<<AB, 256, 0, stream>>>(atomA, dpk[11], Wo2b, nullptr, nullptr, atomB, NATOM);
  final_out_kernel<<<(NATOM + 255)/256, 256, 0, stream>>>(atomB, Wfin, out_v);
}
